// InterpretableMultiHeadAttention_53068615910026
// MI455X (gfx1250) — compile-verified, matrix operand order checked by reading, not run on hardware
//
#include <hip/hip_runtime.h>
#include <hip/hip_bf16.h>

#define BS 32
#define T  512
#define H  1024
#define NH 16
#define DH 64

typedef __attribute__((ext_vector_type(16))) __bf16        v16bf;
typedef __attribute__((ext_vector_type(8)))  float         v8f;
typedef __attribute__((ext_vector_type(4)))  unsigned int  u32x4;
typedef __attribute__((ext_vector_type(4)))  int           i32x4;
typedef __attribute__((ext_vector_type(8)))  int           i32x8;

#define HAS_TDM  __has_builtin(__builtin_amdgcn_tensor_load_to_lds)
#define HAS_TCNT __has_builtin(__builtin_amdgcn_s_wait_tensorcnt)

union FragU { u32x4 q[2]; v16bf v; };

static __device__ inline v8f wmma_bf16(v16bf a, v16bf b, v8f c) {
    // D = A(16x32 bf16) * B(32x16 bf16) + C(16x16 f32)
    return __builtin_amdgcn_wmma_f32_16x16x32_bf16(
        /*neg_a=*/false, a, /*neg_b=*/false, b,
        /*c_mod=*/(short)0, c, /*reuse_a=*/false, /*reuse_b=*/false);
}

// Load a 16x32 bf16 fragment in the CDNA5 16-bit A/B VGPR layout from a
// row-major array (row stride = ld elements). lanes 0-15: row=l16, K={0..7,16..23};
// lanes 16-31: row=l16, K={8..15,24..31}. Two contiguous b128 loads per lane.
static __device__ inline v16bf load_frag(const __hip_bfloat16* base, int row, int ld,
                                         int kOff, int lane) {
    const int l16 = lane & 15;
    const int hi  = lane >> 4;
    const __hip_bfloat16* p = base + (size_t)(row + l16) * ld + kOff + hi * 8;
    FragU f;
    f.q[0] = *(const u32x4*)(p);
    f.q[1] = *(const u32x4*)(p + 16);
    return f.v;
}

#if HAS_TDM
// Issue a TDM 2D tile load (tile_d0 x tile_d1 of 2-byte elements) into LDS.
// D# group0: count=1, lds_addr, global_addr, type=2.
// D# group1: data_size=1 (2B), tensor dims = tile dims, dim0 stride in elements.
// Toolchain uses the 6-arg builtin form: (g0, g1, g2, g3, g_extra, cpol).
static __device__ inline void tdm_load_2d(unsigned lds_byte_addr, const void* gptr,
                                          unsigned tile_d0, unsigned tile_d1,
                                          unsigned long long stride0_elems) {
    unsigned long long ga = (unsigned long long)(size_t)gptr;
    u32x4 g0;
    g0[0] = 1u;                                              // count=1 (valid)
    g0[1] = lds_byte_addr;                                   // lds_addr
    g0[2] = (unsigned)(ga & 0xffffffffu);                    // global_addr[31:0]
    g0[3] = (unsigned)((ga >> 32) & 0x1ffffffu) | (2u << 30); // addr[56:32] | type=2
    i32x8 g1;
    g1[0] = (int)(1u << 16);                                 // data_size=1 (2 bytes)
    g1[1] = (int)((tile_d0 & 0xffffu) << 16);                // tensor_dim0 lo16
    g1[2] = (int)(((tile_d0 >> 16) & 0xffffu) |              // tensor_dim0 hi16
                  ((tile_d1 & 0xffffu) << 16));              // tensor_dim1 lo16
    g1[3] = (int)(((tile_d1 >> 16) & 0xffffu) |              // tensor_dim1 hi16
                  ((tile_d0 & 0xffffu) << 16));              // tile_dim0
    g1[4] = (int)(tile_d1 & 0xffffu);                        // tile_dim1 (tile_dim2=0)
    g1[5] = (int)(stride0_elems & 0xffffffffu);              // tensor_dim0_stride lo32
    g1[6] = (int)((stride0_elems >> 32) & 0xffffu);          // stride0 hi16 (stride1=0)
    g1[7] = 0;
    i32x4 z4 = {0, 0, 0, 0};
    i32x8 z8 = {0, 0, 0, 0, 0, 0, 0, 0};
    __builtin_amdgcn_tensor_load_to_lds(g0, g1, z4, z4, z8, 0);
}
#endif

__global__ void zero_f32_kernel(float* __restrict__ p, int n) {
    for (int i = blockIdx.x * blockDim.x + threadIdx.x; i < n; i += gridDim.x * blockDim.x)
        p[i] = 0.0f;
}

__global__ void cvt_bf16_kernel(const float* __restrict__ s, __hip_bfloat16* __restrict__ d, int n) {
    for (int i = blockIdx.x * blockDim.x + threadIdx.x; i < n; i += gridDim.x * blockDim.x)
        d[i] = __float2bfloat16(s[i]);
}

// W[K][N] (f32) -> WT[N][K] (bf16)
__global__ void tcvt_bf16_kernel(const float* __restrict__ W, __hip_bfloat16* __restrict__ WT,
                                 int K, int N) {
    int total = K * N;
    for (int i = blockIdx.x * blockDim.x + threadIdx.x; i < total; i += gridDim.x * blockDim.x) {
        int k = i / N, n = i - k * N;
        WT[(size_t)n * K + k] = __float2bfloat16(W[i]);
    }
}

__global__ void scale_cvt_bf16_kernel(const float* __restrict__ s, __hip_bfloat16* __restrict__ d,
                                      int n, float sc) {
    for (int i = blockIdx.x * blockDim.x + threadIdx.x; i < n; i += gridDim.x * blockDim.x)
        d[i] = __float2bfloat16(s[i] * sc);
}

// C[M,N] = A[M,K](bf16) @ BT[N,K]^T(bf16) + bias.  8 waves/block, wave -> 16x64 C tile.
// The 64x32 B tile is staged into LDS once per block per k-step via the Tensor Data
// Mover (double-buffered), shared by all 8 waves.
// Cb!=null: store bf16 (vtrans=1 stores V transposed as [b][n][t], T=512).
// Cb==null: store f32 to Cf.
__global__ __launch_bounds__(256) void gemm_bf16_wmma(
    const __hip_bfloat16* __restrict__ A, const __hip_bfloat16* __restrict__ BT,
    const float* __restrict__ bias, __hip_bfloat16* __restrict__ Cb,
    float* __restrict__ Cf, int M, int N, int K, int vtrans) {
    const int tid   = threadIdx.x;
    const int lane  = tid & 31;
    const int wave  = tid >> 5;
    const int tileM = blockIdx.y * 128 + wave * 16;
    const int tileN = blockIdx.x * 64;
    const int l16   = lane & 15;
    const int hi8   = (lane >> 4) * 8;

    __shared__ __hip_bfloat16 ldsB[2][64 * 32];   // double-buffered 64x32 B tile

    v8f acc[4] = {v8f{}, v8f{}, v8f{}, v8f{}};

    // --- stage first B tile ---
#if HAS_TDM
    if (wave == 0) {
        tdm_load_2d((unsigned)(size_t)&ldsB[0][0],
                    BT + (size_t)tileN * K, 32u, 64u, (unsigned long long)K);
    }
#else
    {
        int r = tid >> 2, c = (tid & 3) * 8;
        *(u32x4*)(&ldsB[0][r * 32 + c]) = *(const u32x4*)(BT + (size_t)(tileN + r) * K + c);
    }
#endif

    const int nk = K / 32;
    int buf = 0;
    for (int it = 0; it < nk; ++it) {
        const int kk = it * 32;
        const int more = (it + 1 < nk);

        // --- stage next B tile into the other buffer ---
        if (more) {
#if HAS_TDM
            if (wave == 0) {
                tdm_load_2d((unsigned)(size_t)&ldsB[buf ^ 1][0],
                            BT + (size_t)tileN * K + (kk + 32), 32u, 64u,
                            (unsigned long long)K);
            }
#else
            int r = tid >> 2, c = (tid & 3) * 8;
            *(u32x4*)(&ldsB[buf ^ 1][r * 32 + c]) =
                *(const u32x4*)(BT + (size_t)(tileN + r) * K + (kk + 32) + c);
#endif
        }

#if HAS_TDM && HAS_TCNT
        if (wave == 0) {
            // current tile must be complete; next may still be in flight (in-order TDM)
            if (more) __builtin_amdgcn_s_wait_tensorcnt(1);
            else      __builtin_amdgcn_s_wait_tensorcnt(0);
        }
#endif
        __syncthreads();   // publish current LDS tile to all waves

        __builtin_prefetch(A + (size_t)(tileM + l16) * K + kk + 256, 0, 1);
        v16bf af = load_frag(A, tileM, K, kk, lane);
        const __hip_bfloat16* bl = &ldsB[buf][0];
        v16bf b0 = load_frag(bl,  0, 32, 0, lane);
        v16bf b1 = load_frag(bl, 16, 32, 0, lane);
        v16bf b2 = load_frag(bl, 32, 32, 0, lane);
        v16bf b3 = load_frag(bl, 48, 32, 0, lane);
        acc[0] = wmma_bf16(af, b0, acc[0]);
        acc[1] = wmma_bf16(af, b1, acc[1]);
        acc[2] = wmma_bf16(af, b2, acc[2]);
        acc[3] = wmma_bf16(af, b3, acc[3]);

        __syncthreads();   // all waves done reading before buffer is re-staged
        buf ^= 1;
    }

#pragma unroll
    for (int nt = 0; nt < 4; ++nt) {
        const int n  = tileN + nt * 16 + l16;
        const float bn = bias ? bias[n] : 0.0f;
#pragma unroll
        for (int g = 0; g < 8; ++g) {
            const int r = tileM + g + hi8;
            const float c = acc[nt][g] + bn;
            if (Cb) {
                if (vtrans)  // store as [b][n][t] with t = r % 512
                    Cb[(size_t)(r >> 9) * (DH * T) + (size_t)n * T + (r & (T - 1))] =
                        __float2bfloat16(c);
                else
                    Cb[(size_t)r * N + n] = __float2bfloat16(c);
            } else {
                Cf[(size_t)r * N + n] = c;
            }
        }
    }
}

// Flash attention: one wave per (qt, h, b). Q tile 16x64, causal, online softmax.
__global__ __launch_bounds__(32) void attn_kernel(
    const __hip_bfloat16* __restrict__ Qb, const __hip_bfloat16* __restrict__ Kb,
    const __hip_bfloat16* __restrict__ VTb, float* __restrict__ attn_vec,
    float* __restrict__ mattn) {
    const int lane = threadIdx.x;
    const int qt = blockIdx.x, h = blockIdx.y, b = blockIdx.z;
    const __hip_bfloat16* Qr = Qb + (size_t)b * T * H;
    const __hip_bfloat16* Kr = Kb + (size_t)b * T * H;
    const __hip_bfloat16* Vt = VTb + (size_t)b * DH * T;   // [d][t]
    __shared__ __hip_bfloat16 lP[16 * 32];

    const int l16 = lane & 15;
    const int hi8 = (lane >> 4) * 8;
    const float scale = 0.125f;  // dh^-0.5 = 1/8

    v16bf qa0 = load_frag(Qr, qt * 16, H, h * DH + 0,  lane);
    v16bf qa1 = load_frag(Qr, qt * 16, H, h * DH + 32, lane);

    float m[8], l[8];
    v8f O[4] = {v8f{}, v8f{}, v8f{}, v8f{}};
#pragma unroll
    for (int g = 0; g < 8; ++g) { m[g] = -3.0e38f; l[g] = 0.0f; }

    const int jmax = (qt * 16 + 15) >> 5;  // causal bound, 32-wide k-tiles
    for (int j = 0; j <= jmax; ++j) {
        v8f S0 = {}, S1 = {};
        v16bf k00 = load_frag(Kr, j * 32,      H, h * DH + 0,  lane);
        v16bf k01 = load_frag(Kr, j * 32,      H, h * DH + 32, lane);
        v16bf k10 = load_frag(Kr, j * 32 + 16, H, h * DH + 0,  lane);
        v16bf k11 = load_frag(Kr, j * 32 + 16, H, h * DH + 32, lane);
        S0 = wmma_bf16(qa0, k00, S0);
        S0 = wmma_bf16(qa1, k01, S0);
        S1 = wmma_bf16(qa0, k10, S1);
        S1 = wmma_bf16(qa1, k11, S1);

        const int c0 = j * 32 + l16, c1 = c0 + 16;
#pragma unroll
        for (int g = 0; g < 8; ++g) {
            const int rq = qt * 16 + g + hi8;
            float s0 = (c0 <= rq) ? S0[g] * scale : -3.0e38f;
            float s1 = (c1 <= rq) ? S1[g] * scale : -3.0e38f;
            float mx = fmaxf(s0, s1);
#pragma unroll
            for (int o = 1; o < 16; o <<= 1) mx = fmaxf(mx, __shfl_xor(mx, o, 16));
            const float mnew  = fmaxf(m[g], mx);
            const float alpha = __expf(m[g] - mnew);
            const float p0 = __expf(s0 - mnew);
            const float p1 = __expf(s1 - mnew);
            float rs = p0 + p1;
#pragma unroll
            for (int o = 1; o < 16; o <<= 1) rs += __shfl_xor(rs, o, 16);
            l[g] = l[g] * alpha + rs;
            m[g] = mnew;
            O[0][g] *= alpha; O[1][g] *= alpha; O[2][g] *= alpha; O[3][g] *= alpha;
            const int r = g + hi8;
            lP[r * 32 + l16]      = __float2bfloat16(p0);
            lP[r * 32 + 16 + l16] = __float2bfloat16(p1);
        }
        __syncthreads();  // single-wave WG: LDS stays in order, barrier is a NOP
        v16bf pa = load_frag(lP, 0, 32, 0, lane);
#pragma unroll
        for (int f = 0; f < 4; ++f) {
            v16bf vb = load_frag(Vt, f * 16, T, j * 32, lane);
            O[f] = wmma_bf16(pa, vb, O[f]);
        }
        __syncthreads();
    }

#pragma unroll
    for (int g = 0; g < 8; ++g) {
        const float inv = 1.0f / l[g];
        const int tq = qt * 16 + g + hi8;
#pragma unroll
        for (int f = 0; f < 4; ++f) {
            const int d = f * 16 + l16;
            const float val = O[f][g] * inv;
            attn_vec[(((size_t)b * NH + h) * T + tq) * DH + d] = val;
            atomicAdd(&mattn[((size_t)b * T + tq) * DH + d], val);
        }
    }
}

extern "C" void kernel_launch(void* const* d_in, const int* in_sizes, int n_in,
                              void* d_out, int out_size, void* d_ws, size_t ws_size,
                              hipStream_t stream) {
    const float* x  = (const float*)d_in[0];
    const float* Wq = (const float*)d_in[1];
    const float* bq = (const float*)d_in[2];
    const float* Wk = (const float*)d_in[3];
    const float* bk = (const float*)d_in[4];
    const float* Wv = (const float*)d_in[5];
    const float* bv = (const float*)d_in[6];
    const float* Wo = (const float*)d_in[7];

    float* out      = (float*)d_out;                      // [b,t,1024]
    float* attn_vec = out + (size_t)BS * T * H;           // [b,h,t,64]

    char* ws = (char*)d_ws;
    // workspace layout (bytes), all 16B-aligned
    __hip_bfloat16* xb     = (__hip_bfloat16*)(ws + 0);                  // 32 MB
    __hip_bfloat16* Qbf    = (__hip_bfloat16*)(ws + 33554432ull);        // 32 MB
    __hip_bfloat16* Kbf    = (__hip_bfloat16*)(ws + 67108864ull);        // 32 MB
    __hip_bfloat16* VTb    = (__hip_bfloat16*)(ws + 100663296ull);       // 2  MB
    __hip_bfloat16* WqT    = (__hip_bfloat16*)(ws + 102760448ull);       // 2  MB
    __hip_bfloat16* WkT    = (__hip_bfloat16*)(ws + 104857600ull);       // 2  MB
    __hip_bfloat16* WvT    = (__hip_bfloat16*)(ws + 106954752ull);       // 128 KB
    __hip_bfloat16* WoT    = (__hip_bfloat16*)(ws + 107085824ull);       // 128 KB
    float*          mattn  = (float*)         (ws + 107216896ull);       // 4  MB
    __hip_bfloat16* mattnb = (__hip_bfloat16*)(ws + 111411200ull);       // 2  MB

    const int nx = BS * T * H;       // 16,777,216
    const int nm = BS * T * DH;      // 1,048,576

    zero_f32_kernel<<<dim3(1024), dim3(256), 0, stream>>>(mattn, nm);
    cvt_bf16_kernel<<<dim3(8192), dim3(256), 0, stream>>>(x, xb, nx);
    tcvt_bf16_kernel<<<dim3(4096), dim3(256), 0, stream>>>(Wq, WqT, H, H);
    tcvt_bf16_kernel<<<dim3(4096), dim3(256), 0, stream>>>(Wk, WkT, H, H);
    tcvt_bf16_kernel<<<dim3(256),  dim3(256), 0, stream>>>(Wv, WvT, H, DH);
    tcvt_bf16_kernel<<<dim3(256),  dim3(256), 0, stream>>>(Wo, WoT, DH, H);

    const int M = BS * T;  // 16384
    // Q = x @ Wq + bq  (bf16 out)
    gemm_bf16_wmma<<<dim3(H / 64, M / 128), dim3(256), 0, stream>>>(
        xb, WqT, bq, Qbf, nullptr, M, H, H, 0);
    // K = x @ Wk + bk
    gemm_bf16_wmma<<<dim3(H / 64, M / 128), dim3(256), 0, stream>>>(
        xb, WkT, bk, Kbf, nullptr, M, H, H, 0);
    // V^T = (x @ Wv + bv)^T stored [b][d][t]
    gemm_bf16_wmma<<<dim3(DH / 64, M / 128), dim3(256), 0, stream>>>(
        xb, WvT, bv, VTb, nullptr, M, DH, H, 1);

    // causal flash attention, writes attn_vec + head-sum
    attn_kernel<<<dim3(T / 16, NH, BS), dim3(32), 0, stream>>>(
        Qbf, Kbf, VTb, attn_vec, mattn);

    // head mean -> bf16
    scale_cvt_bf16_kernel<<<dim3(1024), dim3(256), 0, stream>>>(
        mattn, mattnb, nm, 1.0f / (float)NH);

    // out = m_attn @ Wo  (f32 out)
    gemm_bf16_wmma<<<dim3(H / 64, M / 128), dim3(256), 0, stream>>>(
        mattnb, WoT, nullptr, nullptr, out, M, H, DH, 0);
}
